// GNN_node_50955491999994
// MI455X (gfx1250) — matrix operand order repaired and checked by reading, not compiled, not run on hardware
//
#include <hip/hip_runtime.h>
#include <hip/hip_bf16.h>

#define NN      100000      // nodes
#define NE      256000      // edges
#define EMBD    300
#define K1P     320         // EMB padded to mult of 32 (GEMM1 K)
#define N1R     600         // 2*EMB (GEMM1 real N)
#define N1P     640         // padded to mult of 64
#define K2P     640         // 600 padded to mult of 32 (GEMM2 K)
#define N2R     300         // GEMM2 real N
#define N2P     320         // padded to mult of 64
#define NLAYER  5
#define BNEPS   1e-5f

typedef __attribute__((ext_vector_type(16))) __bf16 v16bf;
typedef __attribute__((ext_vector_type(8)))  __bf16 v8bf;
typedef __attribute__((ext_vector_type(8)))  float  v8f;

// ---------- small elementwise kernels ----------

__global__ void k_zero_f32(float* p, long n) {
    long i = (long)blockIdx.x * blockDim.x + threadIdx.x;
    if (i < n) p[i] = 0.f;
}

// combined bond table: ecomb[combo][f] = Wb[f][ea0] + Wb[f][6+ea1], Wb is (300,9)
__global__ void k_ecomb(const float* __restrict__ Wb, float* __restrict__ ecomb) {
    int idx = blockIdx.x * blockDim.x + threadIdx.x;
    if (idx >= 9 * EMBD) return;
    int combo = idx / EMBD, f = idx % EMBD;
    int a = combo / 3, b = combo % 3;
    ecomb[idx] = Wb[f * 9 + a] + Wb[f * 9 + 6 + b];
}

// msg = relu(h[src] + e); agg[dst] += msg
__global__ void k_message(const float* __restrict__ h, const float* __restrict__ ecomb,
                          const int* __restrict__ src, const int* __restrict__ dst,
                          const int* __restrict__ ea, float* __restrict__ agg) {
    long idx = (long)blockIdx.x * blockDim.x + threadIdx.x;
    if (idx >= (long)NE * EMBD) return;
    int e = (int)(idx / EMBD), f = (int)(idx % EMBD);
    int s = src[e], d = dst[e];
    int combo = ea[2 * e] * 3 + ea[2 * e + 1];
    float m = h[(long)s * EMBD + f] + ecomb[combo * EMBD + f];
    m = m > 0.f ? m : 0.f;
    atomicAdd(&agg[(long)d * EMBD + f], m);
}

// A1 = bf16((1+eps)*h + agg), padded K1P columns
__global__ void k_make_a1(const float* __restrict__ h, const float* __restrict__ agg,
                          const float* __restrict__ eps_l, __bf16* __restrict__ out) {
    long idx = (long)blockIdx.x * blockDim.x + threadIdx.x;
    if (idx >= (long)NN * K1P) return;
    int m = (int)(idx / K1P), k = (int)(idx % K1P);
    float v = 0.f;
    if (k < EMBD) v = (1.f + eps_l[0]) * h[(long)m * EMBD + k] + agg[(long)m * EMBD + k];
    out[idx] = (__bf16)v;
}

// pad/convert f32 weights (rows x cols, row-major) to bf16 (rpad x cpad)
__global__ void k_pad_w(const float* __restrict__ W, int rows, int cols,
                        int rpad, int cpad, __bf16* __restrict__ out) {
    long idx = (long)blockIdx.x * blockDim.x + threadIdx.x;
    if (idx >= (long)rpad * cpad) return;
    int r = (int)(idx / cpad), c = (int)(idx % cpad);
    float v = (r < rows && c < cols) ? W[(long)r * cols + c] : 0.f;
    out[idx] = (__bf16)v;
}

// ---------- WMMA GEMM: C[M,Npad] = A[M,K](bf16) x Bt[Npad,K](bf16)^T + bias ----------
// 8 waves/block as 4(M) x 2(N); each wave register-tiles 2(M) x 4(N) WMMA tiles
// (32x64 C). Per K-step: 2 A-frags (4 b128) + 4 B-frags (8 b128) feed 8 WMMAs
// -> 1.5 b128 loads per WMMA.
__global__ void __launch_bounds__(256) k_gemm_bf16(
    const __bf16* __restrict__ A, int lda,
    const __bf16* __restrict__ Bt, int ldb,
    const float* __restrict__ bias, int ncReal,
    float* __restrict__ C, int ldc, int M, int Npad, int K)
{
    const int lane = threadIdx.x & 31;
    const int wid  = threadIdx.x >> 5;
    const int wm   = wid >> 1;               // 0..3 (M)
    const int wn   = wid & 1;                // 0..1 (N)
    const int m0   = blockIdx.y * 128 + wm * 32;
    const int n0   = blockIdx.x * 128 + wn * 64;
    if (m0 >= M || n0 >= Npad) return;       // wave-uniform guards, EXEC stays all-1s
    const int l    = lane & 15;
    const int hi   = lane >> 4;

    // A fragment layout: lane<16 -> row m+l, K k0..k0+15
    //                    lane>=16 -> row m+l, K k0+16..k0+31
    const __bf16* aPtr0 = A + (size_t)(m0 + l) * lda + hi * 16;
    const __bf16* aPtr1 = aPtr0 + (size_t)16 * lda;
    // B fragment layout: lane = column n+l; lane<16 -> K k0..k0+15, lane>=16 -> k0+16..k0+31
    const __bf16* bPtr0 = Bt + (size_t)(n0 + l) * ldb + hi * 16;
    const __bf16* bPtr1 = bPtr0 + (size_t)16 * ldb;
    const __bf16* bPtr2 = bPtr0 + (size_t)32 * ldb;
    const __bf16* bPtr3 = bPtr0 + (size_t)48 * ldb;

    v8f acc00 = {}, acc01 = {}, acc02 = {}, acc03 = {};
    v8f acc10 = {}, acc11 = {}, acc12 = {}, acc13 = {};

    for (int k0 = 0; k0 < K; k0 += 32) {
        v8bf a0lo = *(const v8bf*)(aPtr0 + k0);
        v8bf a0hi = *(const v8bf*)(aPtr0 + k0 + 8);
        v8bf a1lo = *(const v8bf*)(aPtr1 + k0);
        v8bf a1hi = *(const v8bf*)(aPtr1 + k0 + 8);
        v16bf b0  = *(const v16bf*)(bPtr0 + k0);
        v16bf b1  = *(const v16bf*)(bPtr1 + k0);
        v16bf b2  = *(const v16bf*)(bPtr2 + k0);
        v16bf b3  = *(const v16bf*)(bPtr3 + k0);
        v16bf a0, a1;
#pragma unroll
        for (int i = 0; i < 8; ++i) {
            a0[i] = a0lo[i]; a0[8 + i] = a0hi[i];
            a1[i] = a1lo[i]; a1[8 + i] = a1hi[i];
        }
        acc00 = __builtin_amdgcn_wmma_f32_16x16x32_bf16(false, a0, false, b0, (short)0, acc00, false, false);
        acc01 = __builtin_amdgcn_wmma_f32_16x16x32_bf16(false, a0, false, b1, (short)0, acc01, false, false);
        acc02 = __builtin_amdgcn_wmma_f32_16x16x32_bf16(false, a0, false, b2, (short)0, acc02, false, false);
        acc03 = __builtin_amdgcn_wmma_f32_16x16x32_bf16(false, a0, false, b3, (short)0, acc03, false, false);
        acc10 = __builtin_amdgcn_wmma_f32_16x16x32_bf16(false, a1, false, b0, (short)0, acc10, false, false);
        acc11 = __builtin_amdgcn_wmma_f32_16x16x32_bf16(false, a1, false, b1, (short)0, acc11, false, false);
        acc12 = __builtin_amdgcn_wmma_f32_16x16x32_bf16(false, a1, false, b2, (short)0, acc12, false, false);
        acc13 = __builtin_amdgcn_wmma_f32_16x16x32_bf16(false, a1, false, b3, (short)0, acc13, false, false);
    }

    // C tile layout: VGPR r -> row (hi*8 + r), column l (per 16x16 tile)
    const int c0 = n0 + l, c1 = c0 + 16, c2 = c0 + 32, c3 = c0 + 48;
    const float bv0 = (c0 < ncReal) ? bias[c0] : 0.f;
    const float bv1 = (c1 < ncReal) ? bias[c1] : 0.f;
    const float bv2 = (c2 < ncReal) ? bias[c2] : 0.f;
    const float bv3 = (c3 < ncReal) ? bias[c3] : 0.f;
    float* r0 = C + (size_t)(m0 + hi * 8) * ldc;
    float* r1 = C + (size_t)(m0 + 16 + hi * 8) * ldc;
#pragma unroll
    for (int r = 0; r < 8; ++r) {
        r0[(size_t)r * ldc + c0] = acc00[r] + bv0;
        r0[(size_t)r * ldc + c1] = acc01[r] + bv1;
        r0[(size_t)r * ldc + c2] = acc02[r] + bv2;
        r0[(size_t)r * ldc + c3] = acc03[r] + bv3;
        r1[(size_t)r * ldc + c0] = acc10[r] + bv0;
        r1[(size_t)r * ldc + c1] = acc11[r] + bv1;
        r1[(size_t)r * ldc + c2] = acc12[r] + bv2;
        r1[(size_t)r * ldc + c3] = acc13[r] + bv3;
    }
}

// ---------- batch-norm over rows ----------

#define ROWCHUNK 1000
__global__ void k_colstat(const float* __restrict__ C, int ldc, int nc,
                          float* __restrict__ sum, float* __restrict__ sumsq) {
    int c = blockIdx.x * blockDim.x + threadIdx.x;
    if (c >= nc) return;
    long r0 = (long)blockIdx.y * ROWCHUNK;
    long r1 = r0 + ROWCHUNK; if (r1 > NN) r1 = NN;
    float s = 0.f, ss = 0.f;
    for (long r = r0; r < r1; ++r) {
        float v = C[r * ldc + c];
        s += v; ss += v * v;
    }
    atomicAdd(&sum[c], s);
    atomicAdd(&sumsq[c], ss);
}

// colA = g*rsqrt(var+eps); colB = b - mean*colA  =>  bn(x) = x*colA + colB
__global__ void k_bnparams(const float* __restrict__ sum, const float* __restrict__ sumsq,
                           const float* __restrict__ g, const float* __restrict__ b,
                           int nc, float* __restrict__ colA, float* __restrict__ colB) {
    int c = blockIdx.x * blockDim.x + threadIdx.x;
    if (c >= nc) return;
    float mean = sum[c] * (1.f / NN);
    float var  = sumsq[c] * (1.f / NN) - mean * mean;
    float rs   = rsqrtf(var + BNEPS);
    float a    = g[c] * rs;
    colA[c] = a;
    colB[c] = b[c] - mean * a;
}

// relu(bn(C)) -> bf16, padded to ocols (GEMM2 A operand)
__global__ void k_norm_relu_bf16(const float* __restrict__ C, int ldc, int nc,
                                 const float* __restrict__ colA, const float* __restrict__ colB,
                                 __bf16* __restrict__ out, int ocols) {
    long idx = (long)blockIdx.x * blockDim.x + threadIdx.x;
    if (idx >= (long)NN * ocols) return;
    int m = (int)(idx / ocols), c = (int)(idx % ocols);
    float v = 0.f;
    if (c < nc) {
        v = C[(long)m * ldc + c] * colA[c] + colB[c];
        v = v > 0.f ? v : 0.f;
    }
    out[idx] = (__bf16)v;
}

// bn(C) (+relu) -> f32 (next-layer h or final output)
__global__ void k_norm_f32(const float* __restrict__ C, int ldc, int nc,
                           const float* __restrict__ colA, const float* __restrict__ colB,
                           float* __restrict__ out, int doRelu) {
    long idx = (long)blockIdx.x * blockDim.x + threadIdx.x;
    if (idx >= (long)NN * nc) return;
    int m = (int)(idx / nc), c = (int)(idx % nc);
    float v = C[(long)m * ldc + c] * colA[c] + colB[c];
    if (doRelu) v = v > 0.f ? v : 0.f;
    out[idx] = v;
}

// ---------- host ----------

static inline long cdivl(long a, long b) { return (a + b - 1) / b; }

extern "C" void kernel_launch(void* const* d_in, const int* in_sizes, int n_in,
                              void* d_out, int out_size, void* d_ws, size_t ws_size,
                              hipStream_t stream) {
    const float* x      = (const float*)d_in[0];
    const float* W_bond = (const float*)d_in[1];   // (5,300,9)
    const float* eps    = (const float*)d_in[2];   // (5,)
    const float* W1     = (const float*)d_in[3];   // (5,600,300)
    const float* b1     = (const float*)d_in[4];   // (5,600)
    const float* bn1_g  = (const float*)d_in[5];
    const float* bn1_b  = (const float*)d_in[6];
    const float* W2     = (const float*)d_in[7];   // (5,300,600)
    const float* b2     = (const float*)d_in[8];   // (5,300)
    const float* bn_g   = (const float*)d_in[9];
    const float* bn_b   = (const float*)d_in[10];
    const int*   eidx   = (const int*)d_in[11];    // (2,E)
    const int*   eattr  = (const int*)d_in[12];    // (E,2)
    float*       out    = (float*)d_out;

    const int* srcIdx = eidx;
    const int* dstIdx = eidx + NE;

    // ---- carve workspace (256B aligned regions) ----
    char* p = (char*)d_ws;
    size_t off = 0;
    auto carve = [&](size_t bytes) -> char* {
        char* q = p + off;
        off += (bytes + 255) & ~(size_t)255;
        return q;
    };
    float*  hbuf  = (float*) carve((size_t)NN * EMBD * 4);        // ping h buffer
    float*  Cbuf  = (float*) carve((size_t)NN * N1P * 4);         // agg / C1 / C2 (aliased)
    __bf16* Abuf  = (__bf16*)carve((size_t)NN * K2P * 2);         // A1 / A2 (aliased)
    float*  ecomb = (float*) carve((size_t)9 * EMBD * 4);
    __bf16* W1b   = (__bf16*)carve((size_t)N1P * K1P * 2);
    __bf16* W2b   = (__bf16*)carve((size_t)N2P * K2P * 2);
    float*  stats = (float*) carve((size_t)4 * K2P * 4);          // sum|sumsq|colA|colB
    float* sum   = stats;
    float* sumsq = stats + K2P;
    float* colA  = stats + 2 * K2P;
    float* colB  = stats + 3 * K2P;
    (void)ws_size; (void)in_sizes; (void)n_in; (void)out_size;

    float* agg = Cbuf;   // alias: agg dead before C1 written

    for (int l = 0; l < NLAYER; ++l) {
        const float* h = (l == 0) ? x : hbuf;

        // 1) edge table + zero agg + scatter messages
        k_ecomb<<<dim3(cdivl(9 * EMBD, 256)), dim3(256), 0, stream>>>(
            W_bond + (size_t)l * EMBD * 9, ecomb);
        k_zero_f32<<<dim3(cdivl((long)NN * EMBD, 256)), dim3(256), 0, stream>>>(
            agg, (long)NN * EMBD);
        k_message<<<dim3(cdivl((long)NE * EMBD, 256)), dim3(256), 0, stream>>>(
            h, ecomb, srcIdx, dstIdx, eattr, agg);

        // 2) A1 = bf16((1+eps)h + agg), K padded to 320
        k_make_a1<<<dim3(cdivl((long)NN * K1P, 256)), dim3(256), 0, stream>>>(
            h, agg, eps + l, Abuf);

        // 3) W1 (600x300, already [out][in] => Bt layout) -> bf16 padded 640x320
        k_pad_w<<<dim3(cdivl((long)N1P * K1P, 256)), dim3(256), 0, stream>>>(
            W1 + (size_t)l * N1R * EMBD, N1R, EMBD, N1P, K1P, W1b);

        // 4) GEMM1: C1[N, 640] = A1 x W1b^T + b1
        k_gemm_bf16<<<dim3(cdivl(N1P, 128), cdivl(NN, 128)), dim3(256), 0, stream>>>(
            Abuf, K1P, W1b, K1P, b1 + (size_t)l * N1R, N1R, Cbuf, N1P, NN, N1P, K1P);

        // 5) BN1 stats + params; relu(bn1) -> bf16 A2 (K padded to 640)
        k_zero_f32<<<dim3(cdivl(2 * K2P, 256)), dim3(256), 0, stream>>>(sum, 2 * K2P);
        k_colstat<<<dim3(cdivl(N1R, 256), cdivl(NN, ROWCHUNK)), dim3(256), 0, stream>>>(
            Cbuf, N1P, N1R, sum, sumsq);
        k_bnparams<<<dim3(cdivl(N1R, 256)), dim3(256), 0, stream>>>(
            sum, sumsq, bn1_g + (size_t)l * N1R, bn1_b + (size_t)l * N1R, N1R, colA, colB);
        k_norm_relu_bf16<<<dim3(cdivl((long)NN * K2P, 256)), dim3(256), 0, stream>>>(
            Cbuf, N1P, N1R, colA, colB, Abuf, K2P);

        // 6) W2 (300x600 = [out][in]) -> bf16 padded 320x640
        k_pad_w<<<dim3(cdivl((long)N2P * K2P, 256)), dim3(256), 0, stream>>>(
            W2 + (size_t)l * N2R * N1R, N2R, N1R, N2P, K2P, W2b);

        // 7) GEMM2: C2[N, 320] = A2 x W2b^T + b2
        k_gemm_bf16<<<dim3(cdivl(N2P, 128), cdivl(NN, 128)), dim3(256), 0, stream>>>(
            Abuf, K2P, W2b, K2P, b2 + (size_t)l * N2R, N2R, Cbuf, N2P, NN, N2P, K2P);

        // 8) BN2 + (relu if l<4) -> next h (or final output)
        k_zero_f32<<<dim3(cdivl(2 * K2P, 256)), dim3(256), 0, stream>>>(sum, 2 * K2P);
        k_colstat<<<dim3(cdivl(N2R, 256), cdivl(NN, ROWCHUNK)), dim3(256), 0, stream>>>(
            Cbuf, N2P, N2R, sum, sumsq);
        k_bnparams<<<dim3(cdivl(N2R, 256)), dim3(256), 0, stream>>>(
            sum, sumsq, bn_g + (size_t)l * N2R, bn_b + (size_t)l * N2R, N2R, colA, colB);
        float* dst = (l == NLAYER - 1) ? out : hbuf;
        k_norm_f32<<<dim3(cdivl((long)NN * N2R, 256)), dim3(256), 0, stream>>>(
            Cbuf, N2P, N2R, colA, colB, dst, (l < NLAYER - 1) ? 1 : 0);
    }
}
